// AttentionLayer_69475390980556
// MI455X (gfx1250) — matrix operand order repaired and checked by reading, not compiled, not run on hardware
//
#include <hip/hip_runtime.h>

// Problem constants (match reference)
#define B_  16
#define TQ  512
#define TK  1024
#define DD  512

typedef __attribute__((ext_vector_type(16))) __bf16 v16bf;
typedef __attribute__((ext_vector_type(8)))  float  v8f;

union ABfrag { unsigned u[8]; v16bf v; };

// ---------------------------------------------------------------------------
// CDNA5 async global->LDS staging (ASYNCcnt path), with safe fallback
// ---------------------------------------------------------------------------
#if defined(__has_builtin)
#if __has_builtin(__builtin_amdgcn_global_load_async_to_lds_b128) && \
    __has_builtin(__builtin_amdgcn_s_wait_asynccnt)
#define HAS_ASYNC_LDS 1
#endif
#endif

#ifdef HAS_ASYNC_LDS
// Parameter type per hipcc diagnostic: '__attribute__((__vector_size__(4 * sizeof(int)))) int *'
typedef int b128_t __attribute__((vector_size(16)));
__device__ __forceinline__ void cp16_g2l(const void* g, void* l) {
    __builtin_amdgcn_global_load_async_to_lds_b128((b128_t*)(void*)g, (b128_t*)l, 0, 0);
}
template<int N>
__device__ __forceinline__ void cp_wait_le() {
    __builtin_amdgcn_s_wait_asynccnt(N);
}
#else
__device__ __forceinline__ void cp16_g2l(const void* g, void* l) {
    *(uint4*)l = *(const uint4*)g;
}
template<int N>
__device__ __forceinline__ void cp_wait_le() {}
#endif

// round-to-nearest-even f32 -> bf16 (packed pair)
__device__ __forceinline__ unsigned pack2_bf16(float lo, float hi) {
    unsigned a = __float_as_uint(lo);
    unsigned b = __float_as_uint(hi);
    a += 0x7FFFu + ((a >> 16) & 1u);
    b += 0x7FFFu + ((b >> 16) & 1u);
    return (a >> 16) | (b & 0xFFFF0000u);
}
__device__ __forceinline__ unsigned short bf16_1(float x) {
    unsigned a = __float_as_uint(x);
    a += 0x7FFFu + ((a >> 16) & 1u);
    return (unsigned short)(a >> 16);
}

// A-fragment pair index for 16-bit 16x32 A:
// half h=0 lanes hold K {0..15}; h=1 lanes hold K {16..31}
__device__ __forceinline__ int apair(int v, int h) {
    return v + 8 * h;
}

// ---------------------------------------------------------------------------
// Pass 0: keys f32 -> bf16 (packed), layout unchanged [b, k, d]
// ---------------------------------------------------------------------------
__global__ __launch_bounds__(256)
void k_cvt_keys(const float* __restrict__ in, uint2* __restrict__ out) {
    size_t i = (size_t)blockIdx.x * 256 + threadIdx.x;   // one uint2 = 4 bf16
    const float4 f = ((const float4*)in)[i];
    uint2 o;
    o.x = pack2_bf16(f.x, f.y);
    o.y = pack2_bf16(f.z, f.w);
    out[i] = o;
}

// ---------------------------------------------------------------------------
// Pass 1: qlin_bf16[m, n] = bf16( query[m,:] . Wq[n,:] + bq[n] ), m over B*TQ
// ---------------------------------------------------------------------------
__global__ __launch_bounds__(256)
void k_linear_q(const float* __restrict__ query, const float* __restrict__ Wq,
                const float* __restrict__ bq, unsigned short* __restrict__ qlinb) {
    __shared__ unsigned As[128 * 16];   // 128 rows x 16 bf16-pairs (K=32)
    __shared__ unsigned Bs[64 * 16];    // 64 rows  x 16 bf16-pairs

    const int t    = threadIdx.x;
    const int m0   = blockIdx.y * 128;          // over B*TQ
    const int n0   = blockIdx.x * 64;           // over D
    const int lane = t & 31, wave = t >> 5;
    const int h    = lane >> 4, r = lane & 15;

    v8f acc[4] = {};

    for (int k0 = 0; k0 < DD; k0 += 32) {
        // stage A (f32 -> bf16), coalesced float2 per thread
        #pragma unroll
        for (int i = 0; i < 8; ++i) {
            int idx = i * 256 + t;
            int row = idx >> 4, p = idx & 15;
            const float2 q2 = *(const float2*)(query + (size_t)(m0 + row) * DD + k0 + 2 * p);
            As[idx] = pack2_bf16(q2.x, q2.y);
        }
        // stage B = Wq rows n0..n0+63 (Bt[n][k] = Wq[n][k])
        #pragma unroll
        for (int i = 0; i < 4; ++i) {
            int idx = i * 256 + t;
            int n = idx >> 4, p = idx & 15;
            const float2 w2 = *(const float2*)(Wq + (size_t)(n0 + n) * DD + k0 + 2 * p);
            Bs[idx] = pack2_bf16(w2.x, w2.y);
        }
        if (k0 + 32 < DD) {
            __builtin_prefetch(query + (size_t)(m0 + (t >> 1)) * DD + k0 + 32, 0, 1);
            __builtin_prefetch(Wq + (size_t)(n0 + (t & 63)) * DD + k0 + 32, 0, 1);
        }
        __syncthreads();

        ABfrag a;
        #pragma unroll
        for (int v = 0; v < 8; ++v)
            a.u[v] = As[(wave * 16 + r) * 16 + apair(v, h)];
        #pragma unroll
        for (int nf = 0; nf < 4; ++nf) {
            ABfrag bb;
            #pragma unroll
            for (int v = 0; v < 8; ++v)
                bb.u[v] = Bs[(nf * 16 + r) * 16 + v + 8 * h];
            acc[nf] = __builtin_amdgcn_wmma_f32_16x16x32_bf16(
                false, a.v, false, bb.v, (short)0, acc[nf], false, false);
        }
        __syncthreads();
    }

    #pragma unroll
    for (int nf = 0; nf < 4; ++nf) {
        int col = n0 + nf * 16 + r;
        float bias = bq[col];
        #pragma unroll
        for (int v = 0; v < 8; ++v) {
            int m = m0 + wave * 16 + v + 8 * h;
            qlinb[(size_t)m * DD + col] = bf16_1(acc[nf][v] + bias);
        }
    }
}

// ---------------------------------------------------------------------------
// Pass 2: scores[b,q,k] = (qlin[b,q,:] . keys[b,k,:]) * scale, mask k>=len[b]
//          Double-buffered async pipeline: DMA of tile i+1 overlaps WMMA of i.
// ---------------------------------------------------------------------------
__global__ __launch_bounds__(256)
void k_scores(const unsigned short* __restrict__ qlinb, const unsigned short* __restrict__ keysb,
              const int* __restrict__ lens, const float* __restrict__ scale,
              float* __restrict__ attn) {
    __shared__ unsigned As[2][128 * 16];
    __shared__ unsigned Bs[2][64 * 16];

    const int t    = threadIdx.x;
    const int b    = blockIdx.z;
    const int m0   = blockIdx.y * 128;          // over TQ
    const int n0   = blockIdx.x * 64;           // over TK
    const int lane = t & 31, wave = t >> 5;
    const int h    = lane >> 4, r = lane & 15;

    const unsigned* qu = (const unsigned*)qlinb + (size_t)b * TQ * (DD / 2);
    const unsigned* ku = (const unsigned*)keysb + (size_t)b * TK * (DD / 2);

    // 3 async 16B chunks per thread per K-tile
    auto stage = [&](int buf, int k0) {
        #pragma unroll
        for (int i = 0; i < 2; ++i) {
            int idx = i * 256 + t;
            int row = idx >> 2, c = idx & 3;
            cp16_g2l(qu + (size_t)(m0 + row) * (DD / 2) + (k0 >> 1) + c * 4,
                     &As[buf][row * 16 + c * 4]);
        }
        {
            int row = t >> 2, c = t & 3;
            cp16_g2l(ku + (size_t)(n0 + row) * (DD / 2) + (k0 >> 1) + c * 4,
                     &Bs[buf][row * 16 + c * 4]);
        }
    };

    v8f acc[4] = {};

    stage(0, 0);
    for (int k0 = 0, it = 0; k0 < DD; k0 += 32, ++it) {
        const int cur = it & 1;
        if (k0 + 32 < DD) {
            stage(cur ^ 1, k0 + 32);    // prefetch next tile via async DMA
            cp_wait_le<3>();            // drain current tile (in-order completion)
        } else {
            cp_wait_le<0>();
        }
        __syncthreads();

        ABfrag a;
        #pragma unroll
        for (int v = 0; v < 8; ++v)
            a.u[v] = As[cur][(wave * 16 + r) * 16 + apair(v, h)];
        #pragma unroll
        for (int nf = 0; nf < 4; ++nf) {
            ABfrag bb;
            #pragma unroll
            for (int v = 0; v < 8; ++v)
                bb.u[v] = Bs[cur][(nf * 16 + r) * 16 + v + 8 * h];
            acc[nf] = __builtin_amdgcn_wmma_f32_16x16x32_bf16(
                false, a.v, false, bb.v, (short)0, acc[nf], false, false);
        }
        __syncthreads();
    }

    const int   len = lens[b];
    const float sc  = scale[0];
    const float ninf = -__builtin_inff();

    #pragma unroll
    for (int nf = 0; nf < 4; ++nf) {
        int col = n0 + nf * 16 + r;                 // global k
        bool valid = col < len;
        #pragma unroll
        for (int v = 0; v < 8; ++v) {
            int m = m0 + wave * 16 + v + 8 * h;     // q within batch
            float s = valid ? acc[nf][v] * sc : ninf;
            attn[((size_t)b * TQ + m) * TK + col] = s;
        }
    }
}

// ---------------------------------------------------------------------------
// Pass 3: row softmax over TK; writes f32 attn (output) + bf16 copy (ws)
// ---------------------------------------------------------------------------
__global__ __launch_bounds__(256)
void k_softmax(float* __restrict__ attn, unsigned short* __restrict__ attnb) {
    __shared__ float red[256];
    const int row = blockIdx.x;                 // over B*TQ
    const int t   = threadIdx.x;
    float* p = attn + (size_t)row * TK;

    float x[4];
    float m = -__builtin_inff();
    #pragma unroll
    for (int i = 0; i < 4; ++i) {
        x[i] = p[t + i * 256];
        m = fmaxf(m, x[i]);
    }
    red[t] = m; __syncthreads();
    for (int s = 128; s > 0; s >>= 1) {
        if (t < s) red[t] = fmaxf(red[t], red[t + s]);
        __syncthreads();
    }
    m = red[0]; __syncthreads();

    float sum = 0.f;
    #pragma unroll
    for (int i = 0; i < 4; ++i) {
        x[i] = __expf(x[i] - m);                // exp(-inf) = 0 handles mask
        sum += x[i];
    }
    red[t] = sum; __syncthreads();
    for (int s = 128; s > 0; s >>= 1) {
        if (t < s) red[t] += red[t + s];
        __syncthreads();
    }
    const float inv = 1.0f / red[0];

    #pragma unroll
    for (int i = 0; i < 4; ++i) {
        float a = x[i] * inv;
        p[t + i * 256] = a;
        attnb[(size_t)row * TK + t + i * 256] = bf16_1(a);
    }
}

// ---------------------------------------------------------------------------
// Pass 4: context[b,q,d] = sum_k attn[b,q,k] * keys[b,k,d]
//          Double-buffered async pipeline; key tile staged k-major and
//          transposed during B-fragment assembly via 16-bit LDS reads.
// ---------------------------------------------------------------------------
__global__ __launch_bounds__(256)
void k_context(const unsigned short* __restrict__ attnb, const unsigned short* __restrict__ keysb,
               float* __restrict__ ctx) {
    __shared__ unsigned As[2][128 * 16];
    __shared__ unsigned Kt[2][32 * 32];   // [k=32][d=64] bf16, k-major

    const int t    = threadIdx.x;
    const int b    = blockIdx.z;
    const int m0   = blockIdx.y * 128;          // over TQ
    const int n0   = blockIdx.x * 64;           // over D
    const int lane = t & 31, wave = t >> 5;
    const int h    = lane >> 4, r = lane & 15;

    const unsigned* au = (const unsigned*)attnb + (size_t)b * TQ * (TK / 2);
    const unsigned* ku = (const unsigned*)keysb + (size_t)b * TK * (DD / 2);

    auto stage = [&](int buf, int k0) {
        #pragma unroll
        for (int i = 0; i < 2; ++i) {
            int idx = i * 256 + t;
            int row = idx >> 2, c = idx & 3;
            cp16_g2l(au + (size_t)(m0 + row) * (TK / 2) + (k0 >> 1) + c * 4,
                     &As[buf][row * 16 + c * 4]);
        }
        {
            int kk = t >> 3, c = t & 7;         // 8 x 16B chunks per key row
            cp16_g2l(ku + (size_t)(k0 + kk) * (DD / 2) + (n0 >> 1) + c * 4,
                     &Kt[buf][kk * 32 + c * 4]);
        }
    };

    v8f acc[4] = {};

    stage(0, 0);
    for (int k0 = 0, it = 0; k0 < TK; k0 += 32, ++it) {
        const int cur = it & 1;
        if (k0 + 32 < TK) {
            stage(cur ^ 1, k0 + 32);
            cp_wait_le<3>();
        } else {
            cp_wait_le<0>();
        }
        __syncthreads();

        ABfrag a;
        #pragma unroll
        for (int v = 0; v < 8; ++v)
            a.u[v] = As[cur][(wave * 16 + r) * 16 + apair(v, h)];

        const unsigned short* kt = (const unsigned short*)Kt[cur];   // [32][64]
        #pragma unroll
        for (int nf = 0; nf < 4; ++nf) {
            int n = nf * 16 + r;                // local d column
            ABfrag bb;
            #pragma unroll
            for (int v = 0; v < 8; ++v) {
                int kr = 2 * v + 16 * h;        // local k row
                unsigned lo = kt[kr * 64 + n];
                unsigned hi = kt[(kr + 1) * 64 + n];
                bb.u[v] = lo | (hi << 16);
            }
            acc[nf] = __builtin_amdgcn_wmma_f32_16x16x32_bf16(
                false, a.v, false, bb.v, (short)0, acc[nf], false, false);
        }
        __syncthreads();
    }

    #pragma unroll
    for (int nf = 0; nf < 4; ++nf) {
        int col = n0 + nf * 16 + r;
        #pragma unroll
        for (int v = 0; v < 8; ++v) {
            int m = m0 + wave * 16 + v + 8 * h;
            ctx[((size_t)b * TQ + m) * DD + col] = acc[nf][v];
        }
    }
}

// ---------------------------------------------------------------------------
extern "C" void kernel_launch(void* const* d_in, const int* in_sizes, int n_in,
                              void* d_out, int out_size, void* d_ws, size_t ws_size,
                              hipStream_t stream) {
    const float* query = (const float*)d_in[0];
    const float* keys  = (const float*)d_in[1];
    const int*   lens  = (const int*)d_in[2];
    const float* Wq    = (const float*)d_in[3];
    const float* bq    = (const float*)d_in[4];
    const float* scale = (const float*)d_in[5];

    float* ctx  = (float*)d_out;                                  // B*TQ*D
    float* attn = (float*)d_out + (size_t)B_ * TQ * DD;           // B*TQ*TK

    unsigned short* qlinb = (unsigned short*)d_ws;                // B*TQ*D  bf16 (8 MB)
    unsigned short* attnb = qlinb + (size_t)B_ * TQ * DD;         // B*TQ*TK bf16 (16 MB)
    unsigned short* keysb = attnb + (size_t)B_ * TQ * TK;         // B*TK*D  bf16 (16 MB)

    k_cvt_keys<<<dim3((B_ * TK * DD) / (256 * 4)), 256, 0, stream>>>(keys, (uint2*)keysb);
    k_linear_q<<<dim3(DD / 64, (B_ * TQ) / 128), 256, 0, stream>>>(query, Wq, bq, qlinb);
    k_scores  <<<dim3(TK / 64, TQ / 128, B_), 256, 0, stream>>>(qlinb, keysb, lens, scale, attn);
    k_softmax <<<dim3(B_ * TQ), 256, 0, stream>>>(attn, attnb);
    k_context <<<dim3(DD / 64, TQ / 128, B_), 256, 0, stream>>>(attnb, keysb, ctx);
}
